// GPT_34548716929564
// MI455X (gfx1250) — matrix operand order repaired and checked by reading, not compiled, not run on hardware
//
#include <hip/hip_runtime.h>
#include <hip/hip_bf16.h>

typedef __attribute__((ext_vector_type(16))) _Float16 v16h;
typedef __attribute__((ext_vector_type(8)))  _Float16 v8h;
typedef __attribute__((ext_vector_type(8)))  float    v8f;

#define TT 2048
#define DM 512
#define NH 8
#define DHD 64
#define NL 4
#define D4 2048

__device__ __forceinline__ v16h combine8(v8h lo, v8h hi) {
  v16h r;
#pragma unroll
  for (int i = 0; i < 8; ++i) { r[i] = lo[i]; r[i + 8] = hi[i]; }
  return r;
}

__device__ __forceinline__ v8f wmma_f16(v16h a, v16h b, v8f c) {
  return __builtin_amdgcn_wmma_f32_16x16x32_f16(false, a, false, b, (short)0, c,
                                                false, false);
}

// ---------------- elementwise: x = seq + pos ----------------
__global__ void addpos_kernel(const float* __restrict__ a,
                              const float* __restrict__ p,
                              float* __restrict__ x, int n) {
  int i = blockIdx.x * 256 + threadIdx.x;
  if (i < n) x[i] = a[i] + p[i];
}

// ---- weight convert: f32 [L][K][N] row-major -> f16 [L][N][K] (transposed) ----
__global__ void cvt_wt_kernel(const float* __restrict__ src,
                              _Float16* __restrict__ dst, int Kd, int Nd,
                              long total) {
  long i = (long)blockIdx.x * 256 + threadIdx.x;
  if (i >= total) return;
  long mat = (long)Kd * Nd;
  int l = (int)(i / mat);
  long r = i - (long)l * mat;
  int k = (int)(r / Nd);
  int n = (int)(r % Nd);
  dst[(long)l * mat + (long)n * Kd + k] = (_Float16)src[i];
}

// ---------------- layernorm: f32 [T,DM] -> f16 [T,DM] ----------------
__global__ __launch_bounds__(256) void ln_kernel(const float* __restrict__ x,
                                                 const float* __restrict__ w,
                                                 const float* __restrict__ b,
                                                 _Float16* __restrict__ out) {
  const int row = blockIdx.x;
  const int tid = threadIdx.x;
  const float* xr = x + (size_t)row * DM;
  float v0 = xr[tid], v1 = xr[tid + 256];
  __shared__ float s1[256], s2[256];
  s1[tid] = v0 + v1;
  s2[tid] = v0 * v0 + v1 * v1;
  __syncthreads();
  for (int off = 128; off > 0; off >>= 1) {
    if (tid < off) { s1[tid] += s1[tid + off]; s2[tid] += s2[tid + off]; }
    __syncthreads();
  }
  const float mu = s1[0] * (1.0f / DM);
  const float var = s2[0] * (1.0f / DM) - mu * mu;
  const float rs = rsqrtf(var + 1e-5f);
  _Float16* o = out + (size_t)row * DM;
  o[tid]       = (_Float16)((v0 - mu) * rs * w[tid] + b[tid]);
  o[tid + 256] = (_Float16)((v1 - mu) * rs * w[tid + 256] + b[tid + 256]);
}

// ---------------- WMMA GEMM: C[M,N] = A[M,K](f16) @ WT[N,K](f16)^T + bias ----
// mode 0: store f16 [M,N]       mode 1: out32[M,N] += val (residual)
// mode 2: gelu -> f16 [M,N]     mode 3: store f32 [M,N]
// mode 4: store f16 transposed [N,M]
__global__ __launch_bounds__(256) void gemm_wmma(
    const _Float16* __restrict__ A, const _Float16* __restrict__ WT,
    const float* __restrict__ bias, _Float16* __restrict__ out16,
    float* __restrict__ out32, int M, int N, int K, int mode) {
  const int lane = threadIdx.x & 31;
  const int wave = threadIdx.x >> 5;
  const int m0 = blockIdx.y * 128 + wave * 16;
  const int n0 = blockIdx.x * 64;
  const int lmod = lane & 15;
  const int lhalf = (lane & 16) ? 8 : 0;   // unused; A also uses kgrp
  const int kgrp  = (lane & 16) ? 16 : 0;  // B: K offset per lane group

  v8f acc[4];
#pragma unroll
  for (int t = 0; t < 4; ++t)
#pragma unroll
    for (int g = 0; g < 8; ++g) acc[t][g] = 0.0f;

  const _Float16* arow = A + (size_t)(m0 + lmod) * K;
  for (int kk = 0; kk < K; kk += 32) {
    v16h a = *(const v16h*)(arow + kk + kgrp);
#pragma unroll
    for (int t = 0; t < 4; ++t) {
      const _Float16* bp = WT + (size_t)(n0 + t * 16 + lmod) * K + kk + kgrp;
      v16h b = *(const v16h*)bp;
      acc[t] = wmma_f16(a, b, acc[t]);
    }
  }

  const int rbase = m0 + ((lane & 16) ? 8 : 0);
#pragma unroll
  for (int t = 0; t < 4; ++t) {
    const int col = n0 + t * 16 + lmod;
    const float bv = bias ? bias[col] : 0.0f;
#pragma unroll
    for (int g = 0; g < 8; ++g) {
      const int r = rbase + g;
      float val = acc[t][g] + bv;
      if (mode == 0) {
        out16[(size_t)r * N + col] = (_Float16)val;
      } else if (mode == 1) {
        size_t idx = (size_t)r * N + col;
        out32[idx] = out32[idx] + val;
      } else if (mode == 2) {
        val = 0.5f * val * (1.0f + erff(val * 0.70710678118654752f));
        out16[(size_t)r * N + col] = (_Float16)val;
      } else if (mode == 3) {
        out32[(size_t)r * N + col] = val;
      } else {  // mode 4: transposed f16 store
        out16[(size_t)col * M + r] = (_Float16)val;
      }
    }
  }
}

// ------- flash attention: one wave per (head, 16-query tile), no masking -----
// Q,K: f16 [T,DM] row-major.  VT: f16 [DM,T] (head-major columns).  O: f16 [T,DM]
__global__ __launch_bounds__(32) void flash_attn(const _Float16* __restrict__ Q,
                                                 const _Float16* __restrict__ Kt,
                                                 const _Float16* __restrict__ VT,
                                                 _Float16* __restrict__ O) {
  const int qtiles = TT / 16;
  const int head = blockIdx.x / qtiles;
  const int q0 = (blockIdx.x % qtiles) * 16;
  const int lane = threadIdx.x & 31;
  const int lmod = lane & 15;
  const int lhalf = (lane & 16) ? 8 : 0;
  const int kgrp  = (lane & 16) ? 16 : 0;
  const int hb = head * DHD;

  // Q A-fragments for the two K=32 chunks of DH=64
  const _Float16* qp = Q + (size_t)(q0 + lmod) * DM + hb;
  v16h aq0 = *(const v16h*)(qp + kgrp);
  v16h aq1 = *(const v16h*)(qp + 32 + kgrp);

  float mrun[8], lrun[8];
  v8f ov[4];
#pragma unroll
  for (int g = 0; g < 8; ++g) { mrun[g] = -3.0e38f; lrun[g] = 0.0f; }
#pragma unroll
  for (int t = 0; t < 4; ++t)
#pragma unroll
    for (int g = 0; g < 8; ++g) ov[t][g] = 0.0f;

  __shared__ __align__(32) _Float16 pb[16 * 32];
  const float scale = 0.125f;  // 1/sqrt(64)
  const int rloc = (lane & 16) ? 8 : 0;

  for (int k0 = 0; k0 < TT; k0 += 32) {
    v8f s0, s1;
#pragma unroll
    for (int g = 0; g < 8; ++g) { s0[g] = 0.0f; s1[g] = 0.0f; }
    {
      const _Float16* kp = Kt + (size_t)(k0 + lmod) * DM + hb + kgrp;
      s0 = wmma_f16(aq0, *(const v16h*)kp, s0);
      s0 = wmma_f16(aq1, *(const v16h*)(kp + 32), s0);
      kp = Kt + (size_t)(k0 + 16 + lmod) * DM + hb + kgrp;
      s1 = wmma_f16(aq0, *(const v16h*)kp, s1);
      s1 = wmma_f16(aq1, *(const v16h*)(kp + 32), s1);
    }
    // online softmax, rows live in width-16 lane groups
#pragma unroll
    for (int g = 0; g < 8; ++g) {
      float p0 = s0[g] * scale, p1 = s1[g] * scale;
      float tm = fmaxf(p0, p1);
      tm = fmaxf(tm, __shfl_xor(tm, 1, 16));
      tm = fmaxf(tm, __shfl_xor(tm, 2, 16));
      tm = fmaxf(tm, __shfl_xor(tm, 4, 16));
      tm = fmaxf(tm, __shfl_xor(tm, 8, 16));
      float mnew = fmaxf(mrun[g], tm);
      float alpha = __expf(mrun[g] - mnew);
      mrun[g] = mnew;
      p0 = __expf(p0 - mnew);
      p1 = __expf(p1 - mnew);
      float rs = p0 + p1;
      rs += __shfl_xor(rs, 1, 16);
      rs += __shfl_xor(rs, 2, 16);
      rs += __shfl_xor(rs, 4, 16);
      rs += __shfl_xor(rs, 8, 16);
      lrun[g] = lrun[g] * alpha + rs;
#pragma unroll
      for (int t = 0; t < 4; ++t) ov[t][g] *= alpha;
      pb[(rloc + g) * 32 + lmod] = (_Float16)p0;
      pb[(rloc + g) * 32 + 16 + lmod] = (_Float16)p1;
    }
    __syncthreads();
    const _Float16* pp = pb + lmod * 32;
    v16h pa = *(const v16h*)(pp + kgrp);
    __syncthreads();
#pragma unroll
    for (int t = 0; t < 4; ++t) {
      const _Float16* vp = VT + (size_t)(hb + t * 16 + lmod) * TT + k0 + kgrp;
      ov[t] = wmma_f16(pa, *(const v16h*)vp, ov[t]);
    }
  }

#pragma unroll
  for (int g = 0; g < 8; ++g) {
    const float inv = 1.0f / lrun[g];
    const size_t rbase = (size_t)(q0 + rloc + g) * DM + hb;
#pragma unroll
    for (int t = 0; t < 4; ++t)
      O[rbase + t * 16 + lmod] = (_Float16)(ov[t][g] * inv);
  }
}

// ============================ host orchestration ============================
extern "C" void kernel_launch(void* const* d_in, const int* in_sizes, int n_in,
                              void* d_out, int out_size, void* d_ws,
                              size_t ws_size, hipStream_t stream) {
  const float* seq  = (const float*)d_in[0];
  const float* pos  = (const float*)d_in[1];
  const float* Wq   = (const float*)d_in[2];
  const float* bq   = (const float*)d_in[3];
  const float* Wk   = (const float*)d_in[4];
  const float* bk   = (const float*)d_in[5];
  const float* Wv   = (const float*)d_in[6];
  const float* bv   = (const float*)d_in[7];
  const float* Wp   = (const float*)d_in[8];
  const float* bp   = (const float*)d_in[9];
  const float* W1   = (const float*)d_in[10];
  const float* b1   = (const float*)d_in[11];
  const float* W2   = (const float*)d_in[12];
  const float* b2   = (const float*)d_in[13];
  const float* ln1w = (const float*)d_in[14];
  const float* ln1b = (const float*)d_in[15];
  const float* ln2w = (const float*)d_in[16];
  const float* ln2b = (const float*)d_in[17];
  const float* lnfw = (const float*)d_in[18];
  const float* lnfb = (const float*)d_in[19];
  const float* Wh   = (const float*)d_in[20];

  char* ws = (char*)d_ws;
  size_t off = 0;
  auto alloc = [&](size_t bytes) -> char* {
    char* p = ws + off;
    off += (bytes + 255) & ~(size_t)255;
    return p;
  };
  float*    x     = (float*)alloc((size_t)TT * DM * 4);
  _Float16* h16   = (_Float16*)alloc((size_t)TT * DM * 2);
  _Float16* q16   = (_Float16*)alloc((size_t)TT * DM * 2);
  _Float16* k16   = (_Float16*)alloc((size_t)TT * DM * 2);
  _Float16* vT16  = (_Float16*)alloc((size_t)TT * DM * 2);
  _Float16* o16   = (_Float16*)alloc((size_t)TT * DM * 2);
  _Float16* mid16 = (_Float16*)alloc((size_t)TT * D4 * 2);
  _Float16* wq16  = (_Float16*)alloc((size_t)NL * DM * DM * 2);
  _Float16* wk16  = (_Float16*)alloc((size_t)NL * DM * DM * 2);
  _Float16* wv16  = (_Float16*)alloc((size_t)NL * DM * DM * 2);
  _Float16* wp16  = (_Float16*)alloc((size_t)NL * DM * DM * 2);
  _Float16* w116  = (_Float16*)alloc((size_t)NL * DM * D4 * 2);
  _Float16* w216  = (_Float16*)alloc((size_t)NL * D4 * DM * 2);
  _Float16* wh16  = (_Float16*)alloc((size_t)DM * DM * 2);

  // weight conversion f32 [K,N] -> f16 [N,K] per layer
  {
    long tot = (long)NL * DM * DM;
    int blocks = (int)((tot + 255) / 256);
    cvt_wt_kernel<<<blocks, 256, 0, stream>>>(Wq, wq16, DM, DM, tot);
    cvt_wt_kernel<<<blocks, 256, 0, stream>>>(Wk, wk16, DM, DM, tot);
    cvt_wt_kernel<<<blocks, 256, 0, stream>>>(Wv, wv16, DM, DM, tot);
    cvt_wt_kernel<<<blocks, 256, 0, stream>>>(Wp, wp16, DM, DM, tot);
    long tot2 = (long)NL * DM * D4;
    int blocks2 = (int)((tot2 + 255) / 256);
    cvt_wt_kernel<<<blocks2, 256, 0, stream>>>(W1, w116, DM, D4, tot2);
    cvt_wt_kernel<<<blocks2, 256, 0, stream>>>(W2, w216, D4, DM, tot2);
    long tot3 = (long)DM * DM;
    cvt_wt_kernel<<<(int)((tot3 + 255) / 256), 256, 0, stream>>>(Wh, wh16, DM,
                                                                 DM, tot3);
  }

  addpos_kernel<<<(TT * DM + 255) / 256, 256, 0, stream>>>(seq, pos, x,
                                                           TT * DM);

  dim3 gD(DM / 64, TT / 128);
  dim3 gU(D4 / 64, TT / 128);
  for (int i = 0; i < NL; ++i) {
    const size_t wo = (size_t)i * DM * DM;
    const size_t w1o = (size_t)i * DM * D4;
    ln_kernel<<<TT, 256, 0, stream>>>(x, ln1w + i * DM, ln1b + i * DM, h16);
    gemm_wmma<<<gD, 256, 0, stream>>>(h16, wq16 + wo, bq + i * DM, q16,
                                      nullptr, TT, DM, DM, 0);
    gemm_wmma<<<gD, 256, 0, stream>>>(h16, wk16 + wo, bk + i * DM, k16,
                                      nullptr, TT, DM, DM, 0);
    gemm_wmma<<<gD, 256, 0, stream>>>(h16, wv16 + wo, bv + i * DM, vT16,
                                      nullptr, TT, DM, DM, 4);
    flash_attn<<<NH * (TT / 16), 32, 0, stream>>>(q16, k16, vT16, o16);
    gemm_wmma<<<gD, 256, 0, stream>>>(o16, wp16 + wo, bp + i * DM, nullptr, x,
                                      TT, DM, DM, 1);
    ln_kernel<<<TT, 256, 0, stream>>>(x, ln2w + i * DM, ln2b + i * DM, h16);
    gemm_wmma<<<gU, 256, 0, stream>>>(h16, w116 + w1o, b1 + i * D4, mid16,
                                      nullptr, TT, D4, DM, 2);
    gemm_wmma<<<gD, 256, 0, stream>>>(mid16, w216 + w1o, b2 + i * DM, nullptr,
                                      x, TT, DM, D4, 1);
  }
  ln_kernel<<<TT, 256, 0, stream>>>(x, lnfw, lnfb, h16);
  gemm_wmma<<<gD, 256, 0, stream>>>(h16, wh16, nullptr, nullptr, (float*)d_out,
                                    TT, DM, DM, 3);
  (void)in_sizes; (void)n_in; (void)out_size; (void)ws_size;
}
